// GGNN_11201274708667
// MI455X (gfx1250) — matrix operand order repaired and checked by reading, not compiled, not run on hardware
//
#include <hip/hip_runtime.h>
#include <hip/hip_bf16.h>

// ---------------- problem constants ----------------
#define B_    64
#define N_    128
#define D_    256
#define E_    4
#define M_    (B_ * N_)     // 8192 total rows
#define K2_   (2 * D_)      // 512
#define EN_   (E_ * N_)     // 512
#define ANNOT_ 64
#define STEPS_ 5

typedef __attribute__((ext_vector_type(16))) __bf16 v16bf;
typedef __attribute__((ext_vector_type(8)))  float  v8f;
typedef __attribute__((ext_vector_type(4)))  int    v4i;

// round-to-nearest-even f32 -> bf16
__device__ __forceinline__ unsigned short f2bf(float f) {
    unsigned int u = __builtin_bit_cast(unsigned int, f);
    u += 0x7FFFu + ((u >> 16) & 1u);
    return (unsigned short)(u >> 16);
}

struct Frag32 { v4i lo, hi; };

// Load a 32-byte bf16 fragment as two b128 loads (hioff in bf16 elements)
__device__ __forceinline__ v16bf ldfrag(const unsigned short* p, int hioff) {
    Frag32 f;
    f.lo = *(const v4i*)(p);
    f.hi = *(const v4i*)(p + hioff);
    return __builtin_bit_cast(v16bf, f);
}

// -------------------------------------------------------------------------
// Generic strided-batched bf16 WMMA GEMM:  C[bz] = act(A[bz] @ B[bz]^T' + bias)
//   A  : row-major bf16 (M x K), lda, batch stride sA (elements)
//   Bt : bf16 stored TRANSPOSED (N x K row-major), ldb (= K typically)
//   C  : outmode 0 = f32 row-major, 1 = bf16 row-major, 2 = bf16 transposed
//   act: 0 = none, 1 = sigmoid, 2 = tanh
// Block: 256 thr = 8 waves; block tile 128(M) x 64(N); wave tile 32x32.
// Grid: (N/64, M/128, batches). All dims are exact multiples (no guards).
// -------------------------------------------------------------------------
__global__ __launch_bounds__(256)
void wmma_gemm(const unsigned short* __restrict__ A,  long long sA,
               const unsigned short* __restrict__ Bt, long long sB,
               const float* __restrict__ bias,
               void* __restrict__ C, long long sC,
               int K, int lda, int ldb, int ldc,
               int act, int outmode)
{
    const int bz = blockIdx.z;
    A  += (long long)bz * sA;
    Bt += (long long)bz * sB;

    const int lane = threadIdx.x & 31;
    const int wid  = threadIdx.x >> 5;
    const int wm   = wid & 3;          // 4 waves along M
    const int wn   = wid >> 2;         // 2 waves along N
    const int m0   = blockIdx.y * 128 + wm * 32;
    const int n0   = blockIdx.x * 64  + wn * 32;
    const int lm   = lane & 15;
    const int g    = lane >> 4;

    v8f acc00 = {}, acc01 = {}, acc10 = {}, acc11 = {};

    // A fragment per-lane base: row = m + lm, K-offset g*8 (hi half at +16)
    const unsigned short* Ar0 = A  + (long long)(m0      + lm) * lda + g * 8;
    const unsigned short* Ar1 = A  + (long long)(m0 + 16 + lm) * lda + g * 8;
    // B fragment per-lane base: col (row of Bt) = n + lm, K-offset g*8 (hi half at +16)
    const unsigned short* Br0 = Bt + (long long)(n0      + lm) * ldb + g * 8;
    const unsigned short* Br1 = Bt + (long long)(n0 + 16 + lm) * ldb + g * 8;

    for (int k = 0; k < K; k += 32) {
        v16bf a0 = ldfrag(Ar0 + k, 16);
        v16bf a1 = ldfrag(Ar1 + k, 16);
        v16bf b0 = ldfrag(Br0 + k, 16);
        v16bf b1 = ldfrag(Br1 + k, 16);
        acc00 = __builtin_amdgcn_wmma_f32_16x16x32_bf16(false, a0, false, b0, (short)0, acc00, false, false);
        acc01 = __builtin_amdgcn_wmma_f32_16x16x32_bf16(false, a0, false, b1, (short)0, acc01, false, false);
        acc10 = __builtin_amdgcn_wmma_f32_16x16x32_bf16(false, a1, false, b0, (short)0, acc10, false, false);
        acc11 = __builtin_amdgcn_wmma_f32_16x16x32_bf16(false, a1, false, b1, (short)0, acc11, false, false);
    }

    // ---------------- epilogue ----------------
    v8f accs[2][2] = { { acc00, acc01 }, { acc10, acc11 } };
    #pragma unroll
    for (int i = 0; i < 2; ++i) {
        #pragma unroll
        for (int j = 0; j < 2; ++j) {
            const int gn = n0 + j * 16 + lm;         // global output column
            const float bv = bias ? bias[gn] : 0.0f;
            float outv[8];
            #pragma unroll
            for (int r = 0; r < 8; ++r) {
                float x = accs[i][j][r] + bv;
                if (act == 1)      x = 1.0f / (1.0f + __expf(-x));
                else if (act == 2) x = tanhf(x);
                outv[r] = x;
            }
            const int mrow = m0 + i * 16 + 8 * g;    // first of 8 consecutive rows
            if (outmode == 0) {
                float* Cf = (float*)C + (long long)bz * sC;
                #pragma unroll
                for (int r = 0; r < 8; ++r)
                    Cf[(long long)(mrow + r) * ldc + gn] = outv[r];
            } else if (outmode == 1) {
                unsigned short* Ch = (unsigned short*)C + (long long)bz * sC;
                #pragma unroll
                for (int r = 0; r < 8; ++r)
                    Ch[(long long)(mrow + r) * ldc + gn] = f2bf(outv[r]);
            } else {
                // bf16 transposed: lane owns column gn, rows mrow..mrow+7 contiguous
                unsigned short pk[8];
                #pragma unroll
                for (int r = 0; r < 8; ++r) pk[r] = f2bf(outv[r]);
                unsigned short* Ct = (unsigned short*)C + (long long)bz * sC;
                *(v4i*)(Ct + (long long)gn * ldc + mrow) = *(const v4i*)pk;
            }
        }
    }
}

// ---------------- elementwise helpers ----------------

// f32 -> bf16 row copy with independent leading dims (grid.x*256 covers cols)
__global__ void cvt_rows(const float* __restrict__ src, unsigned short* __restrict__ dst,
                         int src_ld, int dst_ld)
{
    const int c = blockIdx.x * 256 + threadIdx.x;
    const long long r = blockIdx.y;
    dst[r * dst_ld + c] = f2bf(src[r * src_ld + c]);
}

// f32 (K x N) -> bf16 transposed (N x K), batched via grid.z
__global__ void cvt_T(const float* __restrict__ src, unsigned short* __restrict__ dst,
                      int K, int Nn, long long sSrc, long long sDst)
{
    src += (long long)blockIdx.z * sSrc;
    dst += (long long)blockIdx.z * sDst;
    const int k = blockIdx.y;
    const int n = blockIdx.x * 64 + threadIdx.x;
    dst[(long long)n * K + k] = f2bf(src[(long long)k * Nn + n]);
}

// h = prop_state; also seed bf16 copy of h into right half of a_bf
__global__ void init_h(const float* __restrict__ prop, float* __restrict__ h,
                       unsigned short* __restrict__ a_bf)
{
    const int c = threadIdx.x;
    const long long r = blockIdx.x;
    const float v = prop[r * D_ + c];
    h[r * D_ + c] = v;
    a_bf[r * K2_ + D_ + c] = f2bf(v);
}

// right half of a_bf := bf16(r * h)   (forms `joined`, left half already a_in)
__global__ void make_joined(const float* __restrict__ rg, const float* __restrict__ h,
                            unsigned short* __restrict__ a_bf)
{
    const int c = threadIdx.x;
    const long long r = blockIdx.x;
    const long long i = r * D_ + c;
    a_bf[r * K2_ + D_ + c] = f2bf(rg[i] * h[i]);
}

// h = (1-z)*h + z*h_hat; refresh bf16 copy in right half of a_bf
__global__ void update_h(const float* __restrict__ z, const float* __restrict__ hhat,
                         float* __restrict__ h, unsigned short* __restrict__ a_bf)
{
    const int c = threadIdx.x;
    const long long r = blockIdx.x;
    const long long i = r * D_ + c;
    const float zz = z[i];
    const float hn = (1.0f - zz) * h[i] + zz * hhat[i];
    h[i] = hn;
    a_bf[r * K2_ + D_ + c] = f2bf(hn);
}

__device__ __forceinline__ unsigned int pcg(unsigned int v) {
    unsigned int s = v * 747796405u + 2891336453u;
    unsigned int w = ((s >> ((s >> 28) + 4u)) ^ s) * 277803737u;
    return (w >> 22) ^ w;
}

// zlat = eps * exp(0.5*var) + mu  (bf16 for decoder GEMMs); eps ~ N(0,1) hash RNG
__global__ void make_zlat(const float* __restrict__ mu, const float* __restrict__ var,
                          unsigned short* __restrict__ zb)
{
    const int c = threadIdx.x;
    const long long r = blockIdx.x;
    const long long i = r * D_ + c;
    const unsigned int idx = (unsigned int)i;
    const float u1 = ((float)pcg(idx) + 1.0f) * 2.3283064e-10f;      // (0,1]
    const float u2 = (float)pcg(idx ^ 0x9E3779B9u) * 2.3283064e-10f; // [0,1)
    const float eps = __fsqrt_rn(-2.0f * __logf(u1)) * __sinf(6.2831853f * u2);
    const float s = __expf(0.5f * var[i]);
    zb[i] = f2bf(eps * s + mu[i]);
}

// -------------------------------------------------------------------------
extern "C" void kernel_launch(void* const* d_in, const int* in_sizes, int n_in,
                              void* d_out, int out_size, void* d_ws, size_t ws_size,
                              hipStream_t stream)
{
    const float* prop   = (const float*)d_in[0];
    const float* Amat   = (const float*)d_in[1];
    const float* W_in   = (const float*)d_in[2];
    const float* b_in   = (const float*)d_in[3];
    const float* W_r    = (const float*)d_in[4];
    const float* b_r    = (const float*)d_in[5];
    const float* W_z    = (const float*)d_in[6];
    const float* b_z    = (const float*)d_in[7];
    const float* W_t    = (const float*)d_in[8];
    const float* b_t    = (const float*)d_in[9];
    const float* W_mu1  = (const float*)d_in[10];
    const float* b_mu1  = (const float*)d_in[11];
    const float* W_mu2  = (const float*)d_in[12];
    const float* b_mu2  = (const float*)d_in[13];
    const float* W_var1 = (const float*)d_in[14];
    const float* b_var1 = (const float*)d_in[15];
    const float* W_var2 = (const float*)d_in[16];
    const float* b_var2 = (const float*)d_in[17];
    const float* W_d1   = (const float*)d_in[18];
    const float* b_d1   = (const float*)d_in[19];
    const float* W_d2   = (const float*)d_in[20];
    const float* b_d2   = (const float*)d_in[21];

    // ---- workspace carve (256B aligned) ----
    char* base = (char*)d_ws;
    size_t off = 0;
    auto carve = [&](size_t bytes) -> char* {
        char* p = base + off;
        off += (bytes + 255) & ~(size_t)255;
        return p;
    };
    unsigned short* a_bf = (unsigned short*)carve((size_t)M_ * K2_ * 2);      // [a_in | h/r*h] bf16
    unsigned short* isb  = (unsigned short*)carve((size_t)B_ * EN_ * D_ * 2); // in_states, bf16 TRANSPOSED per batch (D x EN)
    unsigned short* Ab   = (unsigned short*)carve((size_t)M_ * EN_ * 2);      // A_in bf16 row-major
    float* h   = (float*)carve((size_t)M_ * D_ * 4);
    float* rb  = (float*)carve((size_t)M_ * D_ * 4);
    float* zbf = (float*)carve((size_t)M_ * D_ * 4);
    float* hh  = (float*)carve((size_t)M_ * D_ * 4);
    unsigned short* xb   = (unsigned short*)carve((size_t)M_ * D_ * 2);       // bf16 intermediate / zlat
    unsigned short* WinT = (unsigned short*)carve((size_t)E_ * D_ * D_ * 2);
    unsigned short* WrT  = (unsigned short*)carve((size_t)D_ * K2_ * 2);
    unsigned short* WzT  = (unsigned short*)carve((size_t)D_ * K2_ * 2);
    unsigned short* WtT  = (unsigned short*)carve((size_t)D_ * K2_ * 2);
    unsigned short* Wm1T = (unsigned short*)carve((size_t)D_ * D_ * 2);
    unsigned short* Wm2T = (unsigned short*)carve((size_t)D_ * D_ * 2);
    unsigned short* Wv1T = (unsigned short*)carve((size_t)D_ * D_ * 2);
    unsigned short* Wv2T = (unsigned short*)carve((size_t)D_ * D_ * 2);
    unsigned short* Wd1T = (unsigned short*)carve((size_t)ANNOT_ * D_ * 2);
    unsigned short* Wd2T = (unsigned short*)carve((size_t)EN_ * D_ * 2);

    // ---- output sections (node, edge, mu, var) ----
    float* out_node = (float*)d_out;
    float* out_edge = out_node + (size_t)M_ * ANNOT_;
    float* out_mu   = out_edge + (size_t)M_ * EN_;
    float* out_var  = out_mu   + (size_t)M_ * D_;

    auto gemm = [&](const unsigned short* Ap, long long sA,
                    const unsigned short* Bt, long long sB,
                    const float* bias, void* C, long long sC,
                    int K, int lda, int ldb, int ldc,
                    int act, int outmode, int gx, int gy, int gz) {
        wmma_gemm<<<dim3(gx, gy, gz), 256, 0, stream>>>(Ap, sA, Bt, sB, bias, C, sC,
                                                        K, lda, ldb, ldc, act, outmode);
    };

    // ---- one-time conversions (per launch; deterministic) ----
    cvt_T<<<dim3(D_ / 64, D_, E_), 64, 0, stream>>>(W_in,   WinT, D_,  D_,  (long long)D_ * D_, (long long)D_ * D_);
    cvt_T<<<dim3(D_ / 64, K2_, 1), 64, 0, stream>>>(W_r,    WrT,  K2_, D_,  0, 0);
    cvt_T<<<dim3(D_ / 64, K2_, 1), 64, 0, stream>>>(W_z,    WzT,  K2_, D_,  0, 0);
    cvt_T<<<dim3(D_ / 64, K2_, 1), 64, 0, stream>>>(W_t,    WtT,  K2_, D_,  0, 0);
    cvt_T<<<dim3(D_ / 64, D_, 1),  64, 0, stream>>>(W_mu1,  Wm1T, D_,  D_,  0, 0);
    cvt_T<<<dim3(D_ / 64, D_, 1),  64, 0, stream>>>(W_mu2,  Wm2T, D_,  D_,  0, 0);
    cvt_T<<<dim3(D_ / 64, D_, 1),  64, 0, stream>>>(W_var1, Wv1T, D_,  D_,  0, 0);
    cvt_T<<<dim3(D_ / 64, D_, 1),  64, 0, stream>>>(W_var2, Wv2T, D_,  D_,  0, 0);
    cvt_T<<<dim3(1, D_, 1),        64, 0, stream>>>(W_d1,   Wd1T, D_,  ANNOT_, 0, 0);
    cvt_T<<<dim3(EN_ / 64, D_, 1), 64, 0, stream>>>(W_d2,   Wd2T, D_,  EN_, 0, 0);
    // A_in = A[:,:,:512] -> bf16 row-major
    cvt_rows<<<dim3(EN_ / 256, M_), 256, 0, stream>>>(Amat, Ab, 2 * EN_, EN_);
    // h = prop_state; seed bf16(h) into a_bf right half
    init_h<<<M_, 256, 0, stream>>>(prop, h, a_bf);

    // ---- propagation steps ----
    for (int s = 0; s < STEPS_; ++s) {
        // in_states[b, e*N+n, :] = h[b,n,:] @ W_in[e] + b_in[e]
        // -> written bf16-TRANSPOSED per batch: isb[b][d][e*N+n]
        for (int e = 0; e < E_; ++e) {
            gemm(a_bf + D_, (long long)N_ * K2_,          // A = bf16(h) inside a_bf, lda=512
                 WinT + (size_t)e * D_ * D_, 0,
                 b_in + (size_t)e * D_,
                 isb + (size_t)e * N_, (long long)EN_ * D_,
                 /*K=*/D_, /*lda=*/K2_, /*ldb=*/D_, /*ldc=*/EN_,
                 /*act=*/0, /*outmode=*/2,
                 D_ / 64, 1, B_);
        }
        // a_in[b] = A_in[b] @ in_states[b]  -> bf16 into left half of a_bf
        gemm(Ab, (long long)N_ * EN_,
             isb, (long long)EN_ * D_,
             nullptr,
             a_bf, (long long)N_ * K2_,
             /*K=*/EN_, /*lda=*/EN_, /*ldb=*/EN_, /*ldc=*/K2_,
             /*act=*/0, /*outmode=*/1,
             D_ / 64, 1, B_);
        // r = sigmoid(a @ W_r + b_r) ; z = sigmoid(a @ W_z + b_z)
        gemm(a_bf, 0, WrT, 0, b_r, rb, 0, K2_, K2_, K2_, D_, 1, 0, D_ / 64, M_ / 128, 1);
        gemm(a_bf, 0, WzT, 0, b_z, zbf, 0, K2_, K2_, K2_, D_, 1, 0, D_ / 64, M_ / 128, 1);
        // joined = [a_in | r*h]
        make_joined<<<M_, 256, 0, stream>>>(rb, h, a_bf);
        // h_hat = tanh(joined @ W_t + b_t)
        gemm(a_bf, 0, WtT, 0, b_t, hh, 0, K2_, K2_, K2_, D_, 2, 0, D_ / 64, M_ / 128, 1);
        // h = (1-z)*h + z*h_hat ; refresh bf16(h)
        update_h<<<M_, 256, 0, stream>>>(zbf, hh, h, a_bf);
    }

    // ---- heads: mu / var ----
    gemm(a_bf + D_, 0, Wm1T, 0, b_mu1, xb, 0, D_, K2_, D_, D_, 2, 1, D_ / 64, M_ / 128, 1);
    gemm(xb, 0, Wm2T, 0, b_mu2, out_mu, 0, D_, D_, D_, D_, 0, 0, D_ / 64, M_ / 128, 1);
    gemm(a_bf + D_, 0, Wv1T, 0, b_var1, xb, 0, D_, K2_, D_, D_, 2, 1, D_ / 64, M_ / 128, 1);
    gemm(xb, 0, Wv2T, 0, b_var2, out_var, 0, D_, D_, D_, D_, 0, 0, D_ / 64, M_ / 128, 1);

    // ---- reparameterize ----
    make_zlat<<<M_, 256, 0, stream>>>(out_mu, out_var, xb);

    // ---- decoders ----
    gemm(xb, 0, Wd1T, 0, b_d1, out_node, 0, D_, D_, D_, ANNOT_, 0, 0, ANNOT_ / 64, M_ / 128, 1);
    gemm(xb, 0, Wd2T, 0, b_d2, out_edge, 0, D_, D_, D_, EN_,    0, 0, EN_ / 64,    M_ / 128, 1);
}
